// LogLeakLIF_89215060673281
// MI455X (gfx1250) — matrix operand order repaired and checked by reading, not compiled, not run on hardware
//
#include <hip/hip_runtime.h>
#include <hip/hip_bf16.h>

typedef __attribute__((ext_vector_type(16))) _Float16 v16h;
typedef __attribute__((ext_vector_type(8)))  float    v8f;

#define T_STEPS 1000
#define BATCH   128
#define NREC    256
#define NIN     256
#define BN      (BATCH * NREC)          // 32768 elements per timestep slab

// ---------------------------------------------------------------------------
// A fragment: 16x32 f16 tile of a row-major f32 matrix [*, 256].
// Per lane (row M = lane&15): halves 0..15 = K = kbase + (hi?16:0) + 0..15
// ---------------------------------------------------------------------------
__device__ inline v16h load_a_frag_f32(const float* __restrict__ row, int kbase, int hi) {
    const float* p0 = row + kbase + (hi ? 16 : 0);
    const float* p1 = p0 + 8;
    v16h a;
#pragma unroll
    for (int i = 0; i < 8; ++i) {
        a[i]     = (_Float16)p0[i];
        a[i + 8] = (_Float16)p1[i];
    }
    return a;
}

// ---------------------------------------------------------------------------
// B fragment: 32x16 f16 tile from pre-transposed weights wT[n][k] (k contiguous).
// Per lane (col N = lane&15): halves 0..15 = K = kbase + (hi?16:0) + 0..15
// -> one contiguous 32-byte read per lane.
// ---------------------------------------------------------------------------
__device__ inline v16h load_b_frag(const _Float16* __restrict__ wT, int n, int kbase, int hi) {
    const _Float16* p = wT + (size_t)n * NREC + kbase + (hi ? 16 : 0);
    v16h b;
#pragma unroll
    for (int i = 0; i < 16; ++i) b[i] = p[i];
    return b;
}

// ---------------------------------------------------------------------------
// Prep: f32 weights [k][n] -> f16 transposed wT[n][k]; zero grid-barrier counter.
// ---------------------------------------------------------------------------
__global__ void prep_kernel(const float* __restrict__ w_in, const float* __restrict__ w_rec,
                            _Float16* __restrict__ wTin, _Float16* __restrict__ wTrec,
                            unsigned* __restrict__ cnt) {
    int idx = blockIdx.x * blockDim.x + threadIdx.x;     // 0 .. 65535
    if (idx == 0) *cnt = 0u;
    if (idx < NIN * NREC) {
        int k = idx >> 8, n = idx & 255;
        wTin [(size_t)n * NIN  + k] = (_Float16)w_in [idx];
        wTrec[(size_t)n * NREC + k] = (_Float16)w_rec[idx];
    }
}

// ---------------------------------------------------------------------------
// Phase 1: ff[t] = x[t] @ w_in for all t in parallel (written into vs region).
// One wave per 16x16 output tile; K=256 in 8 WMMA steps.
// ---------------------------------------------------------------------------
__global__ void ff_gemm_kernel(const float* __restrict__ x, const _Float16* __restrict__ wTin,
                               float* __restrict__ ff_out) {
    int gwave = (blockIdx.x * blockDim.x + threadIdx.x) >> 5;
    int lane  = threadIdx.x & 31;
    int hi    = lane >> 4;
    int ln    = lane & 15;
    const int n_tiles = T_STEPS * (BATCH / 16) * (NREC / 16);   // 128000
    if (gwave >= n_tiles) return;

    int bn = gwave & 15;
    int bm = (gwave >> 4) & 7;
    int t  = gwave >> 7;

    const float* arow = x + ((size_t)t * BATCH + bm * 16 + ln) * NIN;
    int ncol = bn * 16 + ln;

    v8f acc = {};
#pragma unroll
    for (int kb = 0; kb < NIN / 32; ++kb) {
        v16h a = load_a_frag_f32(arow, kb * 32, hi);
        v16h b = load_b_frag(wTin, ncol, kb * 32, hi);
        acc = __builtin_amdgcn_wmma_f32_16x16x32_f16(false, a, false, b, (short)0, acc,
                                                     false, false);
    }

    float* out = ff_out + ((size_t)t * BATCH + bm * 16 + hi * 8) * NREC + bn * 16 + ln;
#pragma unroll
    for (int r = 0; r < 8; ++r) out[(size_t)r * NREC] = acc[r];
}

// ---------------------------------------------------------------------------
// Phase 2: persistent recurrent scan. 16 blocks x 256 threads = 128 waves,
// one 16x16 state tile per wave, grid-barrier between timesteps.
// ---------------------------------------------------------------------------
__global__ void scan_kernel(const float* __restrict__ z0, const float* __restrict__ v0,
                            const float* __restrict__ t0, const _Float16* __restrict__ wTrec,
                            float* __restrict__ zs, float* __restrict__ vs,
                            unsigned* __restrict__ cnt) {
    int lane = threadIdx.x & 31;
    int hi   = lane >> 4;
    int ln   = lane & 15;
    int w    = blockIdx.x * (blockDim.x >> 5) + (threadIdx.x >> 5);   // 0..127
    int bm   = w >> 4;          // batch tile (0..7)
    int bn   = w & 15;          // neuron tile (0..15)
    int col  = bn * 16 + ln;
    int row0 = bm * 16 + hi * 8;

    float vst[8], tst[8], zst[8];
#pragma unroll
    for (int r = 0; r < 8; ++r) {
        size_t idx = (size_t)(row0 + r) * NREC + col;
        vst[r] = v0[idx];
        tst[r] = t0[idx];
        zst[r] = z0[idx];
    }

    const unsigned nblocks = 16u;

    for (int t = 0; t < T_STEPS; ++t) {
        // ---- GEMM: acc = z_{t-1} @ w_rec  (z from d_out zs slab, f32 -> f16) ----
        const float* zsrc = (t == 0) ? z0 : (zs + (size_t)(t - 1) * BN);
        const float* arow = zsrc + (size_t)(bm * 16 + ln) * NREC;
        v8f acc = {};
#pragma unroll
        for (int kb = 0; kb < NREC / 32; ++kb) {
            v16h a = load_a_frag_f32(arow, kb * 32, hi);
            v16h b = load_b_frag(wTrec, col, kb * 32, hi);
            acc = __builtin_amdgcn_wmma_f32_16x16x32_f16(false, a, false, b, (short)0, acc,
                                                         false, false);
        }

        // ---- pointwise neuron update (vs slab currently holds ff = x@w_in) ----
        float* vptr = vs + (size_t)t * BN + (size_t)row0 * NREC + col;
        float* zptr = zs + (size_t)t * BN + (size_t)row0 * NREC + col;
#pragma unroll
        for (int r = 0; r < 8; ++r) {
            float i_in = acc[r] + vptr[(size_t)r * NREC];
            float h    = (i_in != 0.0f) ? 1.0f : 0.0f;
            float tn   = tst[r] + (1.0f - h);
            float v    = vst[r] * (1.0f - zst[r]);
            if (v <= -1.0f) v = -1.0f;                       // clamp at -1
            // cap = (2^|v| - B)/A with A=B=C=1 ; v_exp_f32 is exp2
            float cap = __builtin_amdgcn_exp2f(fabsf(v)) - 1.0f;
            float arg = fminf(h * (tn + 1.0f), cap);         // DT = 1
            float sgn = (v > 0.0f) ? 1.0f : ((v < 0.0f) ? -1.0f : 0.0f);
            // v_log_f32 is log2
            v = v - sgn * __builtin_amdgcn_logf(1e-5f + arg + 1.0f);
            v = v + i_in;
            tn = tn * (1.0f - h);
            float z = (v > 0.5f) ? 1.0f : 0.0f;              // (v-THR)/THR > 0
            vst[r] = v; tst[r] = tn; zst[r] = z;
            vptr[(size_t)r * NREC] = v;                       // overwrite ff with new_v
            zptr[(size_t)r * NREC] = z;                       // spike feeds next step
        }

        // ---- grid barrier: make zs[t] visible to all waves before step t+1 ----
        __threadfence();
        __syncthreads();
        if (threadIdx.x == 0) {
            __hip_atomic_fetch_add(cnt, 1u, __ATOMIC_ACQ_REL, __HIP_MEMORY_SCOPE_AGENT);
            unsigned target = (unsigned)(t + 1) * nblocks;
            while (__hip_atomic_load(cnt, __ATOMIC_ACQUIRE, __HIP_MEMORY_SCOPE_AGENT) < target)
                __builtin_amdgcn_s_sleep(1);
        }
        __syncthreads();
        __builtin_amdgcn_fence(__ATOMIC_ACQUIRE, "agent");
    }
}

// ---------------------------------------------------------------------------
extern "C" void kernel_launch(void* const* d_in, const int* in_sizes, int n_in,
                              void* d_out, int out_size, void* d_ws, size_t ws_size,
                              hipStream_t stream) {
    const float* x     = (const float*)d_in[0];   // [1000,128,256]
    const float* z0    = (const float*)d_in[1];   // [128,256]
    const float* v0    = (const float*)d_in[2];
    const float* t0    = (const float*)d_in[3];
    const float* w_in  = (const float*)d_in[4];   // [256,256]
    const float* w_rec = (const float*)d_in[5];   // [256,256]

    float* zs = (float*)d_out;                    // [1000,128,256]
    float* vs = zs + (size_t)T_STEPS * BN;        // [1000,128,256]

    _Float16* wTin  = (_Float16*)d_ws;                                   // 128 KB
    _Float16* wTrec = (_Float16*)((char*)d_ws + (size_t)NIN * NREC * 2); // 128 KB
    unsigned* cnt   = (unsigned*)((char*)d_ws + (size_t)2 * NIN * NREC * 2);

    // 1) weight convert/transpose + barrier-counter reset
    prep_kernel<<<256, 256, 0, stream>>>(w_in, w_rec, wTin, wTrec, cnt);

    // 2) parallel feed-forward GEMM: ff -> vs slab. 128000 tiles, 8 waves/block.
    ff_gemm_kernel<<<16000, 256, 0, stream>>>(x, wTin, vs);

    // 3) persistent recurrent scan (16 co-resident blocks, software grid barrier)
    scan_kernel<<<16, 256, 0, stream>>>(z0, v0, t0, wTrec, zs, vs, cnt);
}
